// MoEProjector_61323543052999
// MI455X (gfx1250) — matrix operand order repaired and checked by reading, not compiled, not run on hardware
//
#include <hip/hip_runtime.h>
#include <hip/hip_bf16.h>
#include <math.h>

// ---------------------------------------------------------------------------
// MoE (T=8192, D=1024, O=4096, E=4, top-2) on gfx1250:
//   v_wmma_f32_16x16x32_f16 GEMM core + global_load_async_to_lds_b128 staging
//   with double-buffered LDS tiles (ASYNCcnt-tracked, overlaps load/compute).
// ---------------------------------------------------------------------------

typedef _Float16 v16h __attribute__((ext_vector_type(16)));
typedef _Float16 h8   __attribute__((ext_vector_type(8)));
typedef _Float16 h4   __attribute__((ext_vector_type(4)));
typedef float    v8f  __attribute__((ext_vector_type(8)));

static constexpr int T_TOK = 8192;   // B*N tokens
static constexpr int DIM_D = 1024;
static constexpr int DIM_O = 4096;
static constexpr int N_EXP = 4;
static constexpr int CHUNK = 2048;   // token chunk for H buffer

__device__ __forceinline__ v16h combine16(h8 lo, h8 hi) {
  return __builtin_shufflevector(lo, hi, 0,1,2,3,4,5,6,7,8,9,10,11,12,13,14,15);
}

// Low 32 bits of a generic pointer to a __shared__ object = LDS byte address
// (see ISA 10.2 on the LDS aperture).
__device__ __forceinline__ unsigned lds_lo(const void* p) {
  return (unsigned)(unsigned long long)p;
}

// Async copy 16B global -> LDS, tracked by ASYNCcnt (no VGPR transit).
__device__ __forceinline__ void async_copy_b128(unsigned lds_addr,
                                                const void* sbase,
                                                unsigned voff) {
  asm volatile("global_load_async_to_lds_b128 %0, %1, %2 offset:0"
               :: "v"(lds_addr), "v"(voff), "s"(sbase)
               : "memory");
}

__device__ __forceinline__ void wait_async0() {
  asm volatile("s_wait_asynccnt 0x0" ::: "memory");
}

// ---------------- f32 -> f16 elementwise convert (x) -----------------------
__global__ void moe_conv_f16(const float* __restrict__ src,
                             _Float16* __restrict__ dst, int n4) {
  int i = blockIdx.x * blockDim.x + threadIdx.x;
  if (i < n4) {
    float4 v = ((const float4*)src)[i];
    h4 o = { (_Float16)v.x, (_Float16)v.y, (_Float16)v.z, (_Float16)v.w };
    ((h4*)dst)[i] = o;
  }
}

// ------------- batched tiled transpose + convert: dst[c][r] = src[r][c] ----
__global__ void moe_transpose_f16(const float* __restrict__ src,
                                  _Float16* __restrict__ dst,
                                  int R, int C) {
  __shared__ float tile[32][33];
  const size_t bs = (size_t)R * C;
  const float* s = src + (size_t)blockIdx.z * bs;
  _Float16*    d = dst + (size_t)blockIdx.z * bs;
  int c0 = blockIdx.x * 32, r0 = blockIdx.y * 32;
  int tx = threadIdx.x, ty = threadIdx.y;      // 32 x 8
  #pragma unroll
  for (int i = ty; i < 32; i += 8)
    tile[i][tx] = s[(size_t)(r0 + i) * C + (c0 + tx)];
  __syncthreads();
  #pragma unroll
  for (int i = ty; i < 32; i += 8)
    d[(size_t)(c0 + i) * R + (r0 + tx)] = (_Float16)tile[tx][i];
}

// ---------------- router: logits -> softmax -> top-2 -> renorm -------------
__global__ void moe_router(const float* __restrict__ x,
                           const float* __restrict__ Wg,
                           const float* __restrict__ bg,
                           float* __restrict__ gates) {
  int gw   = (blockIdx.x * blockDim.x + threadIdx.x) >> 5;  // wave32 per token
  int lane = threadIdx.x & 31;
  if (gw >= T_TOK) return;
  const float* xr = x + (size_t)gw * DIM_D;
  float a0 = 0.f, a1 = 0.f, a2 = 0.f, a3 = 0.f;
  for (int d = lane; d < DIM_D; d += 32) {
    float xv = xr[d];
    float4 w = *(const float4*)(Wg + (size_t)d * 4);
    a0 += xv * w.x; a1 += xv * w.y; a2 += xv * w.z; a3 += xv * w.w;
  }
  #pragma unroll
  for (int off = 16; off > 0; off >>= 1) {
    a0 += __shfl_down(a0, off, 32);
    a1 += __shfl_down(a1, off, 32);
    a2 += __shfl_down(a2, off, 32);
    a3 += __shfl_down(a3, off, 32);
  }
  if (lane == 0) {
    float l[4] = { a0 + bg[0], a1 + bg[1], a2 + bg[2], a3 + bg[3] };
    float m = fmaxf(fmaxf(l[0], l[1]), fmaxf(l[2], l[3]));
    float e[4];
    #pragma unroll
    for (int i = 0; i < 4; ++i) e[i] = __expf(l[i] - m);
    int i1 = 0;
    #pragma unroll
    for (int i = 1; i < 4; ++i) if (e[i] > e[i1]) i1 = i;   // first max (stable)
    int i2 = -1;
    #pragma unroll
    for (int i = 0; i < 4; ++i)
      if (i != i1 && (i2 < 0 || e[i] > e[i2])) i2 = i;
    float inv = 1.0f / (e[i1] + e[i2]);
    float4 g = { 0.f, 0.f, 0.f, 0.f };
    float* gp = (float*)&g;
    gp[i1] = e[i1] * inv;
    gp[i2] = e[i2] * inv;
    *(float4*)(gates + (size_t)gw * 4) = g;
  }
}

// ------------------------- WMMA GEMM (fused epilogues) ---------------------
// A: [M][K] row-major f16;  B: [N][K] row-major f16 (i.e. transposed weight)
// GELU_OUT=true : Hout[m][n] = (f16) gelu(acc + bias[n])           (exact erf)
// GELU_OUT=false: Cout[m][n] (= or +=) gates[(row0+m)*4+e] * (acc + bias[n])
template <bool GELU_OUT>
__global__ __launch_bounds__(256) void moe_gemm(
    const _Float16* __restrict__ A, const _Float16* __restrict__ B,
    const float* __restrict__ bias, _Float16* __restrict__ Hout,
    float* __restrict__ Cout, const float* __restrict__ gates,
    int M, int N, int K, int expert, int row0, int initFlag) {
  const int tid   = threadIdx.x;
  const int wave  = tid >> 5;
  const int lane  = tid & 31;
  const int waveM = wave >> 2;          // 0..1  -> 64-row patch
  const int waveN = wave & 3;           // 0..3  -> 32-col patch
  const int half  = lane >> 4;          // 0/1 (half-wave)
  const int l16   = lane & 15;
  const int bM = blockIdx.y * 128;
  const int bN = blockIdx.x * 128;

  // double-buffered K-step-32 tiles, rows padded to 40 halves (80B) so the
  // half-wave b128 fragment reads are LDS-bank-conflict-free
  __shared__ __align__(16) _Float16 As[2][128 * 40];
  __shared__ __align__(16) _Float16 Bs[2][128 * 40];

  // per-thread staging slots: 2 chunks of 8 halves for each of A and B
  const int c8a = tid, c8b = tid + 256;                 // 0..511
  const int rowA0 = c8a >> 2, segA0 = c8a & 3;
  const int rowA1 = c8b >> 2, segA1 = c8b & 3;
  const unsigned gOffA0 = (unsigned)((((size_t)(bM + rowA0) * K) + segA0 * 8) * 2);
  const unsigned gOffA1 = (unsigned)((((size_t)(bM + rowA1) * K) + segA1 * 8) * 2);
  const unsigned gOffB0 = (unsigned)((((size_t)(bN + rowA0) * K) + segA0 * 8) * 2);
  const unsigned gOffB1 = (unsigned)((((size_t)(bN + rowA1) * K) + segA1 * 8) * 2);
  const int ldsIdx0 = rowA0 * 40 + segA0 * 8;
  const int ldsIdx1 = rowA1 * 40 + segA1 * 8;

  auto stage = [&](int k0, int buf) {
    const unsigned kB = (unsigned)(k0 * 2);             // byte offset along K
    async_copy_b128(lds_lo(&As[buf][ldsIdx0]), A, gOffA0 + kB);
    async_copy_b128(lds_lo(&As[buf][ldsIdx1]), A, gOffA1 + kB);
    async_copy_b128(lds_lo(&Bs[buf][ldsIdx0]), B, gOffB0 + kB);
    async_copy_b128(lds_lo(&Bs[buf][ldsIdx1]), B, gOffB1 + kB);
  };

  v8f c[4][2];
  #pragma unroll
  for (int mi = 0; mi < 4; ++mi)
    #pragma unroll
    for (int ni = 0; ni < 2; ++ni) c[mi][ni] = (v8f)(0.0f);

  const int NT = K >> 5;                                // k-steps of 32
  stage(0, 0);
  for (int t = 0; t < NT; ++t) {
    const int buf = t & 1;
    wait_async0();            // this wave's tile-t copies have landed in LDS
    __syncthreads();          // everyone's landed; everyone done reading buf^1
    if (t + 1 < NT) stage((t + 1) << 5, buf ^ 1);       // overlap with compute

    v16h af[4], bf[2];
    #pragma unroll
    for (int mi = 0; mi < 4; ++mi) {    // A frag: row=l16, K {h*8..}+{16+h*8..}
      int row = waveM * 64 + mi * 16 + l16;
      h8 lo = *(const h8*)&As[buf][row * 40 + half * 8];
      h8 hi = *(const h8*)&As[buf][row * 40 + 16 + half * 8];
      af[mi] = combine16(lo, hi);
    }
    #pragma unroll
    for (int ni = 0; ni < 2; ++ni) {    // B frag: col=l16, K {h*8..}+{16+h*8..}
      int col = waveN * 32 + ni * 16 + l16;
      h8 lo = *(const h8*)&Bs[buf][col * 40 + half * 8];
      h8 hi = *(const h8*)&Bs[buf][col * 40 + 16 + half * 8];
      bf[ni] = combine16(lo, hi);
    }
    #pragma unroll
    for (int mi = 0; mi < 4; ++mi)
      #pragma unroll
      for (int ni = 0; ni < 2; ++ni)
        c[mi][ni] = __builtin_amdgcn_wmma_f32_16x16x32_f16(
            false, af[mi], false, bf[ni], (short)0, c[mi][ni], false, false);
  }

  // epilogue: C tile element (r,lane) -> row = r + 8*half, col = l16
  #pragma unroll
  for (int mi = 0; mi < 4; ++mi) {
    int rbase = bM + waveM * 64 + mi * 16 + half * 8;
    float gv[8];
    if constexpr (!GELU_OUT) {
      #pragma unroll
      for (int r = 0; r < 8; ++r)
        gv[r] = gates[(size_t)(row0 + rbase + r) * 4 + expert];
    }
    #pragma unroll
    for (int ni = 0; ni < 2; ++ni) {
      int col = bN + waveN * 32 + ni * 16 + l16;
      float bv = bias[col];
      #pragma unroll
      for (int r = 0; r < 8; ++r) {
        float v = c[mi][ni][r] + bv;
        size_t o = (size_t)(rbase + r) * N + col;
        if constexpr (GELU_OUT) {
          float g = 0.5f * v * (1.0f + erff(v * 0.7071067811865475f));
          Hout[o] = (_Float16)g;
        } else {
          float res = gv[r] * v;
          if (initFlag) Cout[o] = res;
          else          Cout[o] += res;
        }
      }
    }
  }
}

// -------------------- LayerNorm (applies gate_scale first) -----------------
__global__ __launch_bounds__(256) void moe_layernorm(
    float* __restrict__ out, const float* __restrict__ w,
    const float* __restrict__ b, const float* __restrict__ gs, int N) {
  const int tid = threadIdx.x;
  const float scale = gs[0];
  float* p = out + (size_t)blockIdx.x * N;
  float s = 0.f, s2 = 0.f;
  for (int i = tid; i < N; i += 256) {
    float v = p[i] * scale;
    s += v; s2 += v * v;
  }
  #pragma unroll
  for (int off = 16; off > 0; off >>= 1) {
    s  += __shfl_down(s,  off, 32);
    s2 += __shfl_down(s2, off, 32);
  }
  __shared__ float sh0[8], sh1[8];
  __shared__ float s_mu, s_inv;
  int lane = tid & 31, wid = tid >> 5;
  if (lane == 0) { sh0[wid] = s; sh1[wid] = s2; }
  __syncthreads();
  if (tid == 0) {
    float ts = 0.f, ts2 = 0.f;
    #pragma unroll
    for (int i = 0; i < 8; ++i) { ts += sh0[i]; ts2 += sh1[i]; }
    float mu  = ts / N;
    float var = ts2 / N - mu * mu;
    s_mu = mu;
    s_inv = rsqrtf(var + 1e-5f);
  }
  __syncthreads();
  float mu = s_mu, inv = s_inv;
  for (int i = tid; i < N; i += 256) {
    float v = p[i] * scale;
    p[i] = (v - mu) * inv * w[i] + b[i];
  }
}

// ---------------------------------------------------------------------------
extern "C" void kernel_launch(void* const* d_in, const int* in_sizes, int n_in,
                              void* d_out, int out_size, void* d_ws, size_t ws_size,
                              hipStream_t stream) {
  const float* x   = (const float*)d_in[0];   // [T, D]
  const float* Wg  = (const float*)d_in[1];   // [D, E]
  const float* bg  = (const float*)d_in[2];   // [E]
  const float* W1  = (const float*)d_in[3];   // [E, D, O]
  const float* b1  = (const float*)d_in[4];   // [E, O]
  const float* W2  = (const float*)d_in[5];   // [E, O, O]
  const float* b2  = (const float*)d_in[6];   // [E, O]
  const float* lnw = (const float*)d_in[7];   // [O]
  const float* lnb = (const float*)d_in[8];   // [O]
  const float* gsc = (const float*)d_in[9];   // [1]
  float* out = (float*)d_out;                 // [T, O] f32

  // workspace layout (all offsets 128KB-aligned)
  char* ws = (char*)d_ws;
  const size_t OFF_GATES = 0;                                       // 128 KB
  const size_t OFF_XH    = OFF_GATES + (size_t)T_TOK * 4 * 4;       // 16 MB
  const size_t OFF_W1H   = OFF_XH  + (size_t)T_TOK * DIM_D * 2;     // 32 MB
  const size_t OFF_W2H   = OFF_W1H + (size_t)N_EXP * DIM_O * DIM_D * 2; // 128 MB
  const size_t OFF_H     = OFF_W2H + (size_t)N_EXP * DIM_O * DIM_O * 2; // 16 MB
  float*    gates = (float*)(ws + OFF_GATES);
  _Float16* Xh    = (_Float16*)(ws + OFF_XH);
  _Float16* W1h   = (_Float16*)(ws + OFF_W1H);   // [E][O][D] = W1^T per expert
  _Float16* W2h   = (_Float16*)(ws + OFF_W2H);   // [E][Q][O] = W2^T per expert
  _Float16* Hbuf  = (_Float16*)(ws + OFF_H);     // [CHUNK][O] f16

  // 1) convert x -> f16
  {
    int n4 = T_TOK * DIM_D / 4;
    moe_conv_f16<<<(n4 + 255) / 256, 256, 0, stream>>>(x, Xh, n4);
  }
  // 2) transpose+convert weights (batched over experts)
  {
    dim3 blk(32, 8);
    dim3 g1(DIM_O / 32, DIM_D / 32, N_EXP);   // W1[e]: [D][O] -> W1h[e]: [O][D]
    moe_transpose_f16<<<g1, blk, 0, stream>>>(W1, W1h, DIM_D, DIM_O);
    dim3 g2(DIM_O / 32, DIM_O / 32, N_EXP);   // W2[e]: [O][Q] -> W2h[e]: [Q][O]
    moe_transpose_f16<<<g2, blk, 0, stream>>>(W2, W2h, DIM_O, DIM_O);
  }
  // 3) router -> dense gates [T,4]
  moe_router<<<T_TOK / 8, 256, 0, stream>>>(x, Wg, bg, gates);

  // 4) expert FFN, chunked over tokens; gate-weighted accumulate into d_out
  dim3 gg(DIM_O / 128, CHUNK / 128);          // (32, 16)
  for (int ch = 0; ch < T_TOK / CHUNK; ++ch) {
    const int row0 = ch * CHUNK;
    for (int e = 0; e < N_EXP; ++e) {
      // H = gelu(Xh_chunk @ W1[e] + b1[e])       M=CHUNK N=O K=D
      moe_gemm<true><<<gg, 256, 0, stream>>>(
          Xh + (size_t)row0 * DIM_D, W1h + (size_t)e * DIM_O * DIM_D,
          b1 + (size_t)e * DIM_O, Hbuf, nullptr, nullptr,
          CHUNK, DIM_O, DIM_D, e, row0, 0);
      // out_chunk (=/+=) g_e ⊙ (H @ W2[e] + b2[e])   M=CHUNK N=O K=O
      moe_gemm<false><<<gg, 256, 0, stream>>>(
          Hbuf, W2h + (size_t)e * DIM_O * DIM_O,
          b2 + (size_t)e * DIM_O, nullptr,
          out + (size_t)row0 * DIM_O, gates,
          CHUNK, DIM_O, DIM_O, e, row0, (e == 0) ? 1 : 0);
    }
  }
  // 5) gate_scale + LayerNorm (in-place on d_out)
  moe_layernorm<<<T_TOK, 256, 0, stream>>>(out, lnw, lnb, gsc, DIM_O);
}
